// GNNModule_5239860101470
// MI455X (gfx1250) — matrix operand order repaired and checked by reading, not compiled, not run on hardware
//
#include <hip/hip_runtime.h>
#include <hip/hip_bf16.h>
#include <math.h>

typedef __attribute__((ext_vector_type(16))) _Float16 v16h;
typedef __attribute__((ext_vector_type(8)))  _Float16 v8h;
typedef __attribute__((ext_vector_type(8)))  float    v8f;

#define NEG_SLOPE 0.2f

static __device__ __forceinline__ unsigned fkey(float f) {
    unsigned u = __float_as_uint(f);
    return (u & 0x80000000u) ? ~u : (u | 0x80000000u);
}
static __device__ __forceinline__ float funkey(unsigned u) {
    unsigned v = (u & 0x80000000u) ? (u & 0x7FFFFFFFu) : ~u;
    return __uint_as_float(v);
}

// ---------------------------------------------------------------- utilities
__global__ void k_cvt_f16(const float* __restrict__ in, _Float16* __restrict__ out, int n) {
    int t = blockIdx.x * blockDim.x + threadIdx.x;
    if (t < n) out[t] = (_Float16)in[t];
}

__global__ void k_relu_cvt_f16(const float* __restrict__ in, _Float16* __restrict__ out, int n) {
    int t = blockIdx.x * blockDim.x + threadIdx.x;
    if (t < n) out[t] = (_Float16)fmaxf(in[t], 0.0f);
}

__global__ void k_fill_f32(float* __restrict__ p, float v, int n) {
    int t = blockIdx.x * blockDim.x + threadIdx.x;
    if (t < n) p[t] = v;
}

__global__ void k_fill_u32(unsigned* __restrict__ p, unsigned v, int n) {
    int t = blockIdx.x * blockDim.x + threadIdx.x;
    if (t < n) p[t] = v;
}

// out[n*D + j] = bias[j]
__global__ void k_bias_fill(float* __restrict__ out, const float* __restrict__ bias,
                            int total, int D) {
    int t = blockIdx.x * blockDim.x + threadIdx.x;
    if (t < total) out[t] = bias[t % D];
}

// --------------------------------------------------- weight fragment packing
// Pack row-major W[K][Nc] (f32) into WMMA B-fragment order (f16):
// Bp[((kt*ntiles + nt)*32 + lane)*16 + j] = W[kt*32 + (lane/16)*8 + (j%8) + (j/8)*16][nt*16 + lane%16]
__global__ void k_pack_b(const float* __restrict__ W, _Float16* __restrict__ Bp,
                         int K, int Nc) {
    int ntiles = Nc >> 4, ktiles = K >> 5;
    int t = blockIdx.x * blockDim.x + threadIdx.x;
    int total = ktiles * ntiles * 32;
    if (t >= total) return;
    int lane = t & 31;
    int tile = t >> 5;
    int nt = tile % ntiles;
    int kt = tile / ntiles;
    int n = nt * 16 + (lane & 15);
    int kbase = kt * 32 + ((lane >> 4) << 3);
    _Float16* dst = Bp + (size_t)t * 16;
#pragma unroll
    for (int j = 0; j < 16; ++j)
        dst[j] = (_Float16)W[(size_t)(kbase + (j & 7) + ((j >> 3) << 4)) * Nc + n];
}

// ------------------------------------------------------------- WMMA GEMM
// C[M][Nc] (f16 out) = A[M][K] (f16, row-major) * Bp (pre-packed fragments).
// One wave computes a 16 x (16*NT) strip: the A fragment is loaded once per
// k-step and reused across NT independent v_wmma_f32_16x16x32_f16 issues.
// M must be a multiple of 16 (M=50000=3125*16) so EXEC is all-ones at WMMA.
template <int NT>
__global__ void k_gemm_wmma(const _Float16* __restrict__ A,
                            const _Float16* __restrict__ Bp,
                            _Float16* __restrict__ C,
                            int M, int K, int Nc) {
    int ntiles  = Nc >> 4;
    int ngroups = ntiles / NT;
    int mtiles  = M >> 4;
    int wave = (int)((blockIdx.x * blockDim.x + threadIdx.x) >> 5);
    int lane = threadIdx.x & 31;
    if (wave >= mtiles * ngroups) return;          // wave-uniform exit
    int mt = wave / ngroups;
    int ng = wave % ngroups;

    int row  = mt * 16 + (lane & 15);              // A row for this lane
    int koff = (lane < 16) ? 0 : 8;                // half-wave K offset
    int ktiles = K >> 5;

    v8f zero = {};
    v8f acc[NT];
#pragma unroll
    for (int i = 0; i < NT; ++i) acc[i] = zero;

    for (int kt = 0; kt < ktiles; ++kt) {
        const _Float16* arow = A + (size_t)row * K + kt * 32 + koff;
        v8h alo = *(const v8h*)(arow);             // K = koff .. koff+7
        v8h ahi = *(const v8h*)(arow + 16);        // K = 16+koff .. 16+koff+7
        v16h a;
#pragma unroll
        for (int j = 0; j < 8; ++j) { a[j] = alo[j]; a[8 + j] = ahi[j]; }
#pragma unroll
        for (int i = 0; i < NT; ++i) {
            int nt = ng * NT + i;
            v16h b = *(const v16h*)(Bp + (((size_t)kt * ntiles + nt) * 32 + lane) * 16);
            acc[i] = __builtin_amdgcn_wmma_f32_16x16x32_f16(
                false, a, false, b, (short)0, acc[i], false, false);
        }
    }

    int m0 = mt * 16 + ((lane < 16) ? 0 : 8);
    int nlo = lane & 15;
#pragma unroll
    for (int i = 0; i < NT; ++i) {
        int n = (ng * NT + i) * 16 + nlo;
#pragma unroll
        for (int r = 0; r < 8; ++r)
            C[(size_t)(m0 + r) * Nc + n] = (_Float16)acc[i][r];
    }
}

// ------------------------------------------------------------- edge kernels
// Fused: logits[e*H+h] = sum_c att[h][c]*leaky(XL[src][h*C+c]+XR[dst][h*C+c]),
// then atomicMax of the order-preserving key into maxk[dst][h].
__global__ void k_edge_logits_max(const _Float16* __restrict__ XL,
                                  const _Float16* __restrict__ XR,
                                  const float* __restrict__ att,
                                  const int* __restrict__ src, const int* __restrict__ dst,
                                  float* __restrict__ logits, unsigned* __restrict__ maxk,
                                  int E, int H, int Cc) {
    int t = blockIdx.x * blockDim.x + threadIdx.x;
    if (t >= E * H) return;
    int e = t / H, h = t % H;
    int s = src[e], d = dst[e];
    const v8h* pl = (const v8h*)(XL + ((size_t)s * H + h) * Cc);
    const v8h* pr = (const v8h*)(XR + ((size_t)d * H + h) * Cc);
    const float* pa = att + (size_t)h * Cc;
    float acc = 0.0f;
    for (int c8 = 0; c8 < (Cc >> 3); ++c8) {
        v8h a = pl[c8];
        v8h b = pr[c8];
#pragma unroll
        for (int j = 0; j < 8; ++j) {
            float v = (float)a[j] + (float)b[j];
            v = (v > 0.0f) ? v : NEG_SLOPE * v;
            acc += pa[c8 * 8 + j] * v;
        }
    }
    logits[t] = acc;
    atomicMax(&maxk[(size_t)d * H + h], fkey(acc));
}

__global__ void k_edge_exp_sum(float* __restrict__ logits, const int* __restrict__ dst,
                               const unsigned* __restrict__ maxk,
                               float* __restrict__ denom, int E, int H) {
    int t = blockIdx.x * blockDim.x + threadIdx.x;
    if (t >= E * H) return;
    int e = t / H, h = t % H;
    float m  = funkey(maxk[(size_t)dst[e] * H + h]);
    float ex = __expf(logits[t] - m);
    logits[t] = ex;                                   // reuse buffer for ex
    atomicAdd(&denom[(size_t)dst[e] * H + h], ex);
}

// out[dst][h*C+c] += (ex/denom[dst][h]) * XL[src][h*C+c]
__global__ void k_edge_aggregate(const _Float16* __restrict__ XL, const float* __restrict__ ex,
                                 const float* __restrict__ denom,
                                 const int* __restrict__ src, const int* __restrict__ dst,
                                 float* __restrict__ out, int E, int H, int Cc) {
    int t = blockIdx.x * blockDim.x + threadIdx.x;
    if (t >= E * H) return;
    int e = t / H, h = t % H;
    int s = src[e], d = dst[e];
    float alpha = ex[t] / denom[(size_t)d * H + h];
    const v8h* pl = (const v8h*)(XL + ((size_t)s * H + h) * Cc);
    float*     po = out + ((size_t)d * H + h) * Cc;
    for (int c8 = 0; c8 < (Cc >> 3); ++c8) {
        v8h a = pl[c8];
#pragma unroll
        for (int j = 0; j < 8; ++j)
            atomicAdd(&po[c8 * 8 + j], alpha * (float)a[j]);
    }
}

// ------------------------------------------------------------- host launcher
static inline int cdiv(long long a, int b) { return (int)((a + b - 1) / b); }

extern "C" void kernel_launch(void* const* d_in, const int* in_sizes, int n_in,
                              void* d_out, int out_size, void* d_ws, size_t ws_size,
                              hipStream_t stream) {
    const int N = 50000, E = 800000, OBS = 128, HID = 64, H1 = 4, ACT = 32;
    const int D1 = H1 * HID;  // 256

    const float* x    = (const float*)d_in[0];
    const int*   ei   = (const int*)d_in[1];
    const float* W1l  = (const float*)d_in[2];
    const float* W1r  = (const float*)d_in[3];
    const float* att1 = (const float*)d_in[4];
    const float* b1   = (const float*)d_in[5];
    const float* W2l  = (const float*)d_in[6];
    const float* W2r  = (const float*)d_in[7];
    const float* att2 = (const float*)d_in[8];
    const float* b2   = (const float*)d_in[9];
    const int* src = ei;
    const int* dst = ei + E;
    float* out = (float*)d_out;

    // workspace carve-up
    char* ws = (char*)d_ws;
    size_t off = 0;
    auto alloc = [&](size_t bytes) -> char* {
        char* p = ws + off;
        off += (bytes + 255) & ~(size_t)255;
        return p;
    };
    _Float16* xh    = (_Float16*)alloc((size_t)N * OBS * 2);
    _Float16* W1l_p = (_Float16*)alloc((size_t)OBS * D1 * 2);
    _Float16* W1r_p = (_Float16*)alloc((size_t)OBS * D1 * 2);
    _Float16* W2l_p = (_Float16*)alloc((size_t)D1 * ACT * 2);
    _Float16* W2r_p = (_Float16*)alloc((size_t)D1 * ACT * 2);
    _Float16* XL1   = (_Float16*)alloc((size_t)N * D1 * 2);
    _Float16* XR1   = (_Float16*)alloc((size_t)N * D1 * 2);
    float*    ex1   = (float*)alloc((size_t)E * H1 * 4);
    unsigned* maxk1 = (unsigned*)alloc((size_t)N * H1 * 4);
    float*    den1  = (float*)alloc((size_t)N * H1 * 4);
    float*    h1    = (float*)alloc((size_t)N * D1 * 4);
    _Float16* hh    = (_Float16*)alloc((size_t)N * D1 * 2);
    _Float16* XL2   = (_Float16*)alloc((size_t)N * ACT * 2);
    _Float16* XR2   = (_Float16*)alloc((size_t)N * ACT * 2);
    float*    ex2   = (float*)alloc((size_t)E * 4);
    unsigned* maxk2 = (unsigned*)alloc((size_t)N * 4);
    float*    den2  = (float*)alloc((size_t)N * 4);
    (void)ws_size; (void)n_in; (void)in_sizes; (void)out_size;

    const int B = 256;

    // ---- layer 1 dense transforms (WMMA, f16 in / f16 out) ----
    k_cvt_f16<<<cdiv((long long)N * OBS, B), B, 0, stream>>>(x, xh, N * OBS);
    k_pack_b<<<cdiv((OBS / 32) * (D1 / 16) * 32, B), B, 0, stream>>>(W1l, W1l_p, OBS, D1);
    k_pack_b<<<cdiv((OBS / 32) * (D1 / 16) * 32, B), B, 0, stream>>>(W1r, W1r_p, OBS, D1);
    k_pack_b<<<cdiv((D1 / 32) * (ACT / 16) * 32, B), B, 0, stream>>>(W2l, W2l_p, D1, ACT);
    k_pack_b<<<cdiv((D1 / 32) * (ACT / 16) * 32, B), B, 0, stream>>>(W2r, W2r_p, D1, ACT);

    {
        long long waves = (long long)(N / 16) * (D1 / 16 / 4);   // NT = 4
        int blocks = cdiv(waves * 32, B);
        k_gemm_wmma<4><<<blocks, B, 0, stream>>>(xh, W1l_p, XL1, N, OBS, D1);
        k_gemm_wmma<4><<<blocks, B, 0, stream>>>(xh, W1r_p, XR1, N, OBS, D1);
    }

    // ---- layer 1 edge softmax + aggregation ----
    k_fill_u32<<<cdiv((long long)N * H1, B), B, 0, stream>>>(maxk1, 0u, N * H1);
    k_fill_f32<<<cdiv((long long)N * H1, B), B, 0, stream>>>(den1, 0.0f, N * H1);
    k_bias_fill<<<cdiv((long long)N * D1, B), B, 0, stream>>>(h1, b1, N * D1, D1);

    int eh1 = E * H1;
    k_edge_logits_max<<<cdiv(eh1, B), B, 0, stream>>>(XL1, XR1, att1, src, dst,
                                                      ex1, maxk1, E, H1, HID);
    k_edge_exp_sum<<<cdiv(eh1, B), B, 0, stream>>>(ex1, dst, maxk1, den1, E, H1);
    k_edge_aggregate<<<cdiv(eh1, B), B, 0, stream>>>(XL1, ex1, den1, src, dst, h1, E, H1, HID);

    // ---- ReLU + convert for layer 2 ----
    k_relu_cvt_f16<<<cdiv((long long)N * D1, B), B, 0, stream>>>(h1, hh, N * D1);

    // ---- layer 2 dense transforms (WMMA) ----
    {
        long long waves = (long long)(N / 16) * (ACT / 16 / 2);  // NT = 2
        int blocks = cdiv(waves * 32, B);
        k_gemm_wmma<2><<<blocks, B, 0, stream>>>(hh, W2l_p, XL2, N, D1, ACT);
        k_gemm_wmma<2><<<blocks, B, 0, stream>>>(hh, W2r_p, XR2, N, D1, ACT);
    }

    // ---- layer 2 edge softmax + aggregation into d_out ----
    k_fill_u32<<<cdiv(N, B), B, 0, stream>>>(maxk2, 0u, N);
    k_fill_f32<<<cdiv(N, B), B, 0, stream>>>(den2, 0.0f, N);
    k_bias_fill<<<cdiv((long long)N * ACT, B), B, 0, stream>>>(out, b2, N * ACT, ACT);

    k_edge_logits_max<<<cdiv(E, B), B, 0, stream>>>(XL2, XR2, att2, src, dst,
                                                    ex2, maxk2, E, 1, ACT);
    k_edge_exp_sum<<<cdiv(E, B), B, 0, stream>>>(ex2, dst, maxk2, den2, E, 1);
    k_edge_aggregate<<<cdiv(E, B), B, 0, stream>>>(XL2, ex2, den2, src, dst, out, E, 1, ACT);
}
